// CGTD_EP_40613210751434
// MI455X (gfx1250) — matrix operand order repaired and checked by reading, not compiled, not run on hardware
//
#include <hip/hip_runtime.h>
#include <stdint.h>

// ---------------------------------------------------------------------------
// Model constants (from the reference)
// ---------------------------------------------------------------------------
#define D_MODEL 768
#define NHEAD   12
#define HD      64
#define NTOK    64
#define BATCH   4

// ---------------------------------------------------------------------------
// WMMA types (CDNA5 / gfx1250, wave32)
// ---------------------------------------------------------------------------
typedef __attribute__((ext_vector_type(8)))  float  v8f;
typedef __attribute__((ext_vector_type(16))) __bf16 v16bf;

union AFrag { v16bf v; uint2 d[4]; };
union BFrag { v16bf v; uint4 q[2]; };
union Ld8   { uint4 q; uint2 d[2]; __bf16 h[8]; };

__device__ __forceinline__ float actf(float x, int act) {
  if (act == 1) return fmaxf(x, 0.f);
  if (act == 2) return x / (1.f + __expf(-x));   // SiLU
  return x;
}

__device__ __forceinline__ v8f wmma_bf16(AFrag a, BFrag b, v8f c) {
  // D = A(16x32 bf16) x B(32x16 bf16) + C(16x16 f32)
  return __builtin_amdgcn_wmma_f32_16x16x32_bf16(false, a.v, false, b.v,
                                                 (short)0, c, false, false);
}

// ---------------------------------------------------------------------------
// Generic bf16 GEMM with fused epilogue.
//   out[M,N] = act( A[M,K] @ B[K,N] + bias[N] + res[M,N] ) * rowScale[M]
// A row-major (lda), B row-major (ldb), outputs row-major (ldc).
// Block tile 64x64, K-step 32, 8 waves; each wave owns 2 WMMA tiles.
// ---------------------------------------------------------------------------
#define BM 64
#define BN 64
#define BK 32
#define LDA_T 36   // padded LDS row stride (halves); multiple of 4 for 8B stores

__global__ __launch_bounds__(256) void k_gemm_bf16(
    const __bf16* __restrict__ A, int lda,
    const __bf16* __restrict__ B, int ldb,
    const float* __restrict__ bias,
    const float* __restrict__ res, int ldres,
    const float* __restrict__ rowScale,
    float* __restrict__ outF,
    __bf16* __restrict__ outH,
    int M, int N, int K, int ldc, int act)
{
  __shared__ __bf16 sA[BM * LDA_T];   // row-major, padded
  __shared__ __bf16 sB[4 * 512];      // 4 column tiles, fragment-swizzled

  const int tid  = threadIdx.x;
  const int lane = tid & 31;
  const int wv   = tid >> 5;
  const int bm   = blockIdx.x * BM;
  const int bn   = blockIdx.y * BN;
  const int tm   = wv & 3;            // row tile of this wave (0..3)
  const int tn0  = (wv >> 2) * 2;     // first of two column tiles

  v8f acc0 = {}, acc1 = {};

  for (int k0 = 0; k0 < K; k0 += BK) {
    { // ---- stage A tile (row-major): thread t -> row t>>2, 8 halves
      int m  = tid >> 2, kc = (tid & 3) * 8;
      int gm = bm + m,  gk = k0 + kc;
      Ld8 t;
      if (gm < M && gk + 7 < K) {
        t.q = *(const uint4*)(A + (size_t)gm * lda + gk);
      } else {
        #pragma unroll
        for (int i = 0; i < 8; ++i)
          t.h[i] = (gm < M && gk + i < K) ? A[(size_t)gm * lda + gk + i]
                                          : (__bf16)0.f;
      }
      *(uint2*)&sA[m * LDA_T + kc]     = t.d[0];
      *(uint2*)&sA[m * LDA_T + kc + 4] = t.d[1];
    }
    { // ---- stage B tile, swizzled to fragment order:
      //      (k,n) -> (n/16)*512 + ((k>>4)&1)*256 + (n&15)*16 + (k&15)
      int k  = tid >> 3, nc = (tid & 7) * 8;
      int gk = k0 + k,  gn = bn + nc;
      Ld8 t;
      if (gk < K && gn + 7 < N) {
        t.q = *(const uint4*)(B + (size_t)gk * ldb + gn);
      } else {
        #pragma unroll
        for (int i = 0; i < 8; ++i)
          t.h[i] = (gk < K && gn + i < N) ? B[(size_t)gk * ldb + gn + i]
                                          : (__bf16)0.f;
      }
      int base = (nc >> 4) * 512 + ((k >> 4) & 1) * 256 + (nc & 15) * 16 + (k & 15);
      #pragma unroll
      for (int i = 0; i < 8; ++i) sB[base + i * 16] = t.h[i];
    }
    // hint next K-tile into cache while the LDS barrier drains (global_prefetch_b8)
    if (k0 + BK < K) {
      __builtin_prefetch(A + (size_t)(bm + (tid >> 2)) * lda + (k0 + BK), 0, 1);
      __builtin_prefetch(B + (size_t)(k0 + BK + (tid >> 3)) * ldb + bn, 0, 1);
    }
    __syncthreads();
    {
      // A fragment: lane<16 needs K{0..15} of row (tm*16+lane),
      //             lane>=16 needs K{16..31} of row (tm*16+lane-16)
      AFrag a;
      int row = tm * 16 + (lane & 15);
      int cb  = (lane >> 4) * 16;
      const __bf16* ap = &sA[row * LDA_T + cb];
      a.d[0] = *(const uint2*)(ap);
      a.d[1] = *(const uint2*)(ap + 4);
      a.d[2] = *(const uint2*)(ap + 8);
      a.d[3] = *(const uint2*)(ap + 12);
      BFrag b0, b1;
      b0.v = *(const v16bf*)&sB[ tn0      * 512 + lane * 16];
      b1.v = *(const v16bf*)&sB[(tn0 + 1) * 512 + lane * 16];
      acc0 = wmma_bf16(a, b0, acc0);
      acc1 = wmma_bf16(a, b1, acc1);
    }
    __syncthreads();
  }

  // ---- epilogue: C layout is VGPR r -> M = r + 8*(lane>=16), N = lane&15
  int rbase = bm + tm * 16 + ((lane >> 4) * 8);
  int c0    = bn + tn0 * 16 + (lane & 15);
  #pragma unroll
  for (int r = 0; r < 8; ++r) {
    int row = rbase + r;
    if (row >= M) continue;
    float rs = rowScale ? rowScale[row] : 1.f;
    #pragma unroll
    for (int t = 0; t < 2; ++t) {
      int col = c0 + t * 16;
      if (col >= N) continue;
      float v = (t == 0) ? acc0[r] : acc1[r];
      if (bias) v += bias[col];
      if (res)  v += res[(size_t)row * ldres + col];
      v = actf(v, act) * rs;
      if (outF) outF[(size_t)row * ldc + col] = v;
      if (outH) outH[(size_t)row * ldc + col] = (__bf16)v;
    }
  }
}

// ---------------------------------------------------------------------------
// Fused per-(batch, head) attention: S = QK^T/8, masked softmax, O = PV.
// Q/K/V/O are bf16 [B*n, D_MODEL] slices at column h*HD. All in LDS.
// ---------------------------------------------------------------------------
__global__ __launch_bounds__(256) void k_attn(
    const __bf16* __restrict__ Q,
    const __bf16* __restrict__ Km,
    const __bf16* __restrict__ V,
    __bf16* __restrict__ O,
    const unsigned char* __restrict__ valid,
    float scale)
{
  __shared__ __bf16 sQ[NTOK * 72];
  __shared__ __bf16 sK[NTOK * 72];
  __shared__ __bf16 sP[NTOK * 72];
  __shared__ __bf16 sV[8 * 512];       // (kstep*4 + coltile) fragment-swizzled
  __shared__ float  sS[NTOK * 65];

  const int tid  = threadIdx.x, lane = tid & 31, wv = tid >> 5;
  const int b    = blockIdx.x / NHEAD;
  const int h    = blockIdx.x % NHEAD;
  const size_t rowOff = (size_t)b * NTOK * D_MODEL + (size_t)h * HD;

  { // stage Q,K row-major (64x64 each): thread -> row t>>2, 16 halves
    int r = tid >> 2, c0 = (tid & 3) * 16;
    const uint4* qs = (const uint4*)(Q  + rowOff + (size_t)r * D_MODEL + c0);
    const uint4* ks = (const uint4*)(Km + rowOff + (size_t)r * D_MODEL + c0);
    *(uint4*)&sQ[r * 72 + c0]     = qs[0];
    *(uint4*)&sQ[r * 72 + c0 + 8] = qs[1];
    *(uint4*)&sK[r * 72 + c0]     = ks[0];
    *(uint4*)&sK[r * 72 + c0 + 8] = ks[1];
  }
  // stage V fragment-swizzled (contraction = key index k)
  for (int it = 0; it < 2; ++it) {
    int idx = tid + it * 256;
    int k = idx >> 3, d0 = (idx & 7) * 8;
    Ld8 t;
    t.q = *(const uint4*)(V + rowOff + (size_t)k * D_MODEL + d0);
    int base = ((k >> 5) * 4 + (d0 >> 4)) * 512 + ((k >> 4) & 1) * 256
             + (d0 & 15) * 16 + (k & 15);
    #pragma unroll
    for (int i = 0; i < 8; ++i) sV[base + i * 16] = t.h[i];
  }
  __syncthreads();

  const int tm = wv & 3, tn0 = (wv >> 2) * 2;

  { // ---- S = Q K^T : B operand = rows of K (contiguous), no transpose needed
    v8f a0 = {}, a1 = {};
    #pragma unroll
    for (int s = 0; s < 2; ++s) {
      AFrag a;
      int row = tm * 16 + (lane & 15);
      int cb  = s * 32 + (lane >> 4) * 16;
      const __bf16* ap = &sQ[row * 72 + cb];
      a.d[0] = *(const uint2*)(ap);
      a.d[1] = *(const uint2*)(ap + 4);
      a.d[2] = *(const uint2*)(ap + 8);
      a.d[3] = *(const uint2*)(ap + 12);
      BFrag b0, b1;
      int n0 = tn0 * 16 + (lane & 15);
      int kb = s * 32 + 16 * (lane >> 4);
      b0.q[0] = *(const uint4*)&sK[ n0       * 72 + kb];
      b0.q[1] = *(const uint4*)&sK[ n0       * 72 + kb + 8];
      b1.q[0] = *(const uint4*)&sK[(n0 + 16) * 72 + kb];
      b1.q[1] = *(const uint4*)&sK[(n0 + 16) * 72 + kb + 8];
      a0 = wmma_bf16(a, b0, a0);
      a1 = wmma_bf16(a, b1, a1);
    }
    int rb = tm * 16 + (lane >> 4) * 8;
    int c0 = tn0 * 16 + (lane & 15);
    #pragma unroll
    for (int r = 0; r < 8; ++r) {
      float s0 = a0[r] * scale, s1 = a1[r] * scale;
      if (!valid[b * NTOK + c0])      s0 = -1e9f;
      if (!valid[b * NTOK + c0 + 16]) s1 = -1e9f;
      sS[(rb + r) * 65 + c0]      = s0;
      sS[(rb + r) * 65 + c0 + 16] = s1;
    }
  }
  __syncthreads();

  if (tid < NTOK) { // softmax, one row per thread
    float mx = -3.4e38f;
    for (int j = 0; j < NTOK; ++j) mx = fmaxf(mx, sS[tid * 65 + j]);
    float sum = 0.f;
    for (int j = 0; j < NTOK; ++j) sum += __expf(sS[tid * 65 + j] - mx);
    float inv = 1.f / sum;
    for (int j = 0; j < NTOK; ++j)
      sP[tid * 72 + j] = (__bf16)(__expf(sS[tid * 65 + j] - mx) * inv);
  }
  __syncthreads();

  { // ---- O = P V
    v8f a0 = {}, a1 = {};
    #pragma unroll
    for (int s = 0; s < 2; ++s) {
      AFrag a;
      int row = tm * 16 + (lane & 15);
      int cb  = s * 32 + (lane >> 4) * 16;
      const __bf16* ap = &sP[row * 72 + cb];
      a.d[0] = *(const uint2*)(ap);
      a.d[1] = *(const uint2*)(ap + 4);
      a.d[2] = *(const uint2*)(ap + 8);
      a.d[3] = *(const uint2*)(ap + 12);
      BFrag b0, b1;
      b0.v = *(const v16bf*)&sV[(s * 4 + tn0)     * 512 + lane * 16];
      b1.v = *(const v16bf*)&sV[(s * 4 + tn0 + 1) * 512 + lane * 16];
      a0 = wmma_bf16(a, b0, a0);
      a1 = wmma_bf16(a, b1, a1);
    }
    int rb = tm * 16 + (lane >> 4) * 8;
    int c0 = tn0 * 16 + (lane & 15);
    #pragma unroll
    for (int r = 0; r < 8; ++r) {
      O[rowOff + (size_t)(rb + r) * D_MODEL + c0]      = (__bf16)a0[r];
      O[rowOff + (size_t)(rb + r) * D_MODEL + c0 + 16] = (__bf16)a1[r];
    }
  }
}

// ---------------------------------------------------------------------------
// LayerNorm over rows of length C (one block per row), optional fused act.
// ---------------------------------------------------------------------------
__global__ __launch_bounds__(256) void k_layernorm(
    const float* __restrict__ x, const float* __restrict__ g,
    const float* __restrict__ b, float* __restrict__ outF,
    __bf16* __restrict__ outH, int C, int act)
{
  __shared__ float red[256];
  const int row = blockIdx.x, tid = threadIdx.x;
  const float* xr = x + (size_t)row * C;
  float s = 0.f;
  for (int c = tid; c < C; c += 256) s += xr[c];
  red[tid] = s; __syncthreads();
  for (int o = 128; o > 0; o >>= 1) { if (tid < o) red[tid] += red[tid + o]; __syncthreads(); }
  float mean = red[0] / C;
  __syncthreads();
  float v = 0.f;
  for (int c = tid; c < C; c += 256) { float d = xr[c] - mean; v += d * d; }
  red[tid] = v; __syncthreads();
  for (int o = 128; o > 0; o >>= 1) { if (tid < o) red[tid] += red[tid + o]; __syncthreads(); }
  float inv = rsqrtf(red[0] / C + 1e-5f);
  for (int c = tid; c < C; c += 256) {
    float y = actf((xr[c] - mean) * inv * g[c] + b[c], act);
    if (outF) outF[(size_t)row * C + c] = y;
    if (outH) outH[(size_t)row * C + c] = (__bf16)y;
  }
}

// ---------------------------------------------------------------------------
// Small elementwise kernels
// ---------------------------------------------------------------------------
__global__ void k_cvt(const float* __restrict__ x, __bf16* __restrict__ y, size_t n) {
  size_t i = (size_t)blockIdx.x * 256 + threadIdx.x;
  if (i < n) y[i] = (__bf16)x[i];
}

__global__ void k_sincos(const int* __restrict__ t, float* __restrict__ out) {
  int i = blockIdx.x * 256 + threadIdx.x;              // BATCH * half
  const int half = D_MODEL / 2;
  if (i >= BATCH * half) return;
  int bq = i / half, j = i % half;
  float f = __expf(-logf(10000.f) * (float)j / (float)half);
  float arg = (float)t[bq] * f;
  out[bq * D_MODEL + j]        = __sinf(arg);
  out[bq * D_MODEL + half + j] = __cosf(arg);
}

__global__ void k_pool_mean(const float* __restrict__ e, float* __restrict__ out) {
  int idx = blockIdx.x * 256 + threadIdx.x;            // B*n*D
  if (idx >= BATCH * NTOK * D_MODEL) return;
  int d = idx % D_MODEL;
  int bi = idx / D_MODEL;
  const float* p = e + (size_t)bi * NTOK * D_MODEL + d;
  float s = 0.f;
  for (int j = 0; j < NTOK; ++j) s += p[(size_t)j * D_MODEL];
  out[idx] = s * (1.f / NTOK);
}

// dense_ez = (0.5*(surf_i+surf_j) + eemb + temb) * mask    (mask is binary)
__global__ void k_edge_tokens(const float* __restrict__ surf,
                              const float* __restrict__ eemb,
                              const float* __restrict__ temb,
                              const float* __restrict__ mask,
                              float* __restrict__ outF, __bf16* __restrict__ outH) {
  size_t idx = (size_t)blockIdx.x * 256 + threadIdx.x;
  size_t total = (size_t)BATCH * NTOK * NTOK * D_MODEL;
  if (idx >= total) return;
  int d = idx % D_MODEL;
  size_t row = idx / D_MODEL;
  int j = row % NTOK;
  size_t bi = row / NTOK;
  int i = bi % NTOK;
  int bb = (int)(bi / NTOK);
  float si = surf[bi * D_MODEL + d];
  float sj = surf[(bi - i + j) * D_MODEL + d];
  float v = (0.5f * (si + sj) + eemb[idx] + temb[bb * D_MODEL + d]) * mask[row];
  if (outF) outF[idx] = v;
  if (outH) outH[idx] = (__bf16)v;
}

__global__ void k_face_init(const float* __restrict__ surf, const float* __restrict__ pooled,
                            const float* __restrict__ temb,
                            float* __restrict__ xenc, float* __restrict__ hface) {
  int idx = blockIdx.x * 256 + threadIdx.x;            // B*n*D
  if (idx >= BATCH * NTOK * D_MODEL) return;
  int d = idx % D_MODEL;
  int bb = idx / (NTOK * D_MODEL);
  xenc[idx]  = surf[idx] + pooled[idx];
  hface[idx] = surf[idx] + temb[bb * D_MODEL + d];
}

// GINE: hl_in = (1+eps)*hf + sum_j relu(hf[j]+he[i,j]) * adj[i,j]
__global__ void k_gine(const float* __restrict__ hf, const float* __restrict__ he,
                       const float* __restrict__ adj, const float* __restrict__ epsPtr,
                       __bf16* __restrict__ outH) {
  int idx = blockIdx.x * 256 + threadIdx.x;            // B*n*D
  if (idx >= BATCH * NTOK * D_MODEL) return;
  int d  = idx % D_MODEL;
  int bi = idx / D_MODEL;
  int i  = bi % NTOK;
  int bb = bi / NTOK;
  const float* her  = he  + ((size_t)bi * NTOK) * D_MODEL + d;
  const float* adjr = adj + (size_t)bi * NTOK;
  const float* hfb  = hf  + ((size_t)bb * NTOK) * D_MODEL + d;
  float acc = 0.f;
  for (int j = 0; j < NTOK; ++j) {
    float m = hfb[(size_t)j * D_MODEL] + her[(size_t)j * D_MODEL];
    acc += fmaxf(m, 0.f) * adjr[j];
  }
  outH[idx] = (__bf16)((1.f + epsPtr[0]) * hf[idx] + acc);
}

__global__ void k_add2(const float* __restrict__ a, const float* __restrict__ b,
                       float* __restrict__ o, size_t n) {
  size_t i = (size_t)blockIdx.x * 256 + threadIdx.x;
  if (i < n) o[i] = a[i] + b[i];
}

// ew input: he + hf_i + hf_j
__global__ void k_edge_pair(const float* __restrict__ hf, const float* __restrict__ he,
                            __bf16* __restrict__ outH) {
  size_t idx = (size_t)blockIdx.x * 256 + threadIdx.x;
  size_t total = (size_t)BATCH * NTOK * NTOK * D_MODEL;
  if (idx >= total) return;
  int d = idx % D_MODEL;
  size_t row = idx / D_MODEL;
  int j = row % NTOK;
  size_t bi = row / NTOK;
  int i = bi % NTOK;
  size_t bj = bi - i + j;
  outH[idx] = (__bf16)(he[idx] + hf[bi * D_MODEL + d] + hf[bj * D_MODEL + d]);
}

// fc_out2 input: concat(dense_ez, eh, fh_i + fh_j) * mask  -> bf16 [.,2304]
__global__ void k_concat(const float* __restrict__ dez, const float* __restrict__ eh,
                         const float* __restrict__ fh, const float* __restrict__ mask,
                         __bf16* __restrict__ outH) {
  size_t idx = (size_t)blockIdx.x * 256 + threadIdx.x;
  size_t total = (size_t)BATCH * NTOK * NTOK * 2304;
  if (idx >= total) return;
  int cC = (int)(idx % 2304);
  size_t row = idx / 2304;
  int j = row % NTOK;
  size_t bi = row / NTOK;
  int i = bi % NTOK;
  size_t bj = bi - i + j;
  float v;
  if (cC < 768)        v = dez[row * 768 + cC];
  else if (cC < 1536)  v = eh[row * 768 + (cC - 768)];
  else                 v = fh[bi * 768 + (cC - 1536)] + fh[bj * 768 + (cC - 1536)];
  outH[idx] = (__bf16)(v * mask[row]);
}

// fc_pool input: ep + (gs_i + gs_j) * mask
__global__ void k_add_ge(const float* __restrict__ ep, const float* __restrict__ gs,
                         const float* __restrict__ mask, __bf16* __restrict__ outH) {
  size_t idx = (size_t)blockIdx.x * 256 + threadIdx.x;
  size_t total = (size_t)BATCH * NTOK * NTOK * D_MODEL;
  if (idx >= total) return;
  int d = idx % D_MODEL;
  size_t row = idx / D_MODEL;
  int j = row % NTOK;
  size_t bi = row / NTOK;
  int i = bi % NTOK;
  size_t bj = bi - i + j;
  outH[idx] = (__bf16)(ep[idx] + (gs[bi * D_MODEL + d] + gs[bj * D_MODEL + d]) * mask[row]);
}

__global__ void k_sym(const float* __restrict__ pred, float* __restrict__ out) {
  size_t idx = (size_t)blockIdx.x * 256 + threadIdx.x;
  size_t total = (size_t)BATCH * NTOK * NTOK * 6;
  if (idx >= total) return;
  int c = (int)(idx % 6);
  size_t row = idx / 6;
  int j = row % NTOK;
  size_t bi = row / NTOK;
  int i = bi % NTOK;
  size_t rowT = (bi - i + j) * NTOK + i;
  out[idx] = 0.5f * (pred[row * 6 + c] + pred[rowT * 6 + c]);
}

// ---------------------------------------------------------------------------
// Host orchestration
// ---------------------------------------------------------------------------
// Flattened parameter-leaf indexing (depth-first, dict insertion order).
enum { IN_EDGEPOS = 0, IN_SURFPOS, IN_SURFZ, IN_T, IN_FMASK, IN_EMASK, P0 = 6 };
#define P_SURFZ  (P0 + 0)
#define P_SURFP  (P0 + 6)
#define P_EDGEP  (P0 + 12)
#define P_TIME   (P0 + 18)
#define P_NET    (P0 + 24)          // 12 layers x 16 leaves
#define P_NETLN  (P_NET + 192)
#define P_GNN    (P_NETLN + 2)      // 6 blocks x 19 leaves
#define P_FCOUT2 (P_GNN + 114)
#define P_FCPOOL (P_FCOUT2 + 6)
#define P_FCOUT  (P_FCPOOL + 6)

struct Ctx {
  hipStream_t stream;
  void* const* in;
  const int* sizes;
  int n_in;
  char* ws; size_t off; size_t cap;
  const __bf16* Wb[512];
  const float* F(int i) const { return (const float*)in[i < n_in ? i : n_in - 1]; }
  const __bf16* W(int i) const { return Wb[i < n_in ? i : n_in - 1]; }
};

template <typename T> static T* walloc(Ctx& c, size_t n) {
  c.off = (c.off + 255) & ~(size_t)255;
  T* p = (T*)(c.ws + c.off);
  c.off += n * sizeof(T);
  return p;
}

static inline unsigned gblk(size_t n) { return (unsigned)((n + 255) / 256); }

static void cvt(Ctx& c, const float* x, __bf16* y, size_t n) {
  k_cvt<<<gblk(n), 256, 0, c.stream>>>(x, y, n);
}

static void gemm(Ctx& c, const __bf16* A, int lda, const __bf16* B, int ldb,
                 const float* bias, const float* res, int ldres,
                 const float* rowScale, float* outF, __bf16* outH,
                 int M, int N, int K, int ldc, int act) {
  dim3 g((M + 63) / 64, (N + 63) / 64);
  k_gemm_bf16<<<g, 256, 0, c.stream>>>(A, lda, B, ldb, bias, res, ldres,
                                       rowScale, outF, outH, M, N, K, ldc, act);
}

// mlp4 = Linear(pi) -> LayerNorm(pi+2,+3) -> SiLU -> Linear(pi+4)
static void mlp4(Ctx& c, const __bf16* A, int M, int din, int dmid, int dout,
                 int pi, const float* res, const float* rowScale,
                 float* outF, __bf16* outH, int ldc) {
  float*  t1 = walloc<float>(c, (size_t)M * dmid);
  gemm(c, A, din, c.W(pi), dmid, c.F(pi + 1), nullptr, 0, nullptr,
       t1, nullptr, M, dmid, din, dmid, 0);
  __bf16* t2 = walloc<__bf16>(c, (size_t)M * dmid);
  k_layernorm<<<M, 256, 0, c.stream>>>(t1, c.F(pi + 2), c.F(pi + 3),
                                       nullptr, t2, dmid, 2 /*SiLU*/);
  gemm(c, t2, dmid, c.W(pi + 4), dout, c.F(pi + 5), res, ldc, rowScale,
       outF, outH, M, dout, dmid, ldc, 0);
}

// MHA on 256 face tokens: q/k/v projections, fused attention, o-projection.
static void mha(Ctx& c, const __bf16* Xh, int pi, const float* res,
                float* outF, __bf16* outH,
                __bf16* qb, __bf16* kb, __bf16* vb, __bf16* ab,
                const unsigned char* valid) {
  const int M = BATCH * NTOK;
  gemm(c, Xh, D_MODEL, c.W(pi + 0), D_MODEL, c.F(pi + 1), nullptr, 0, nullptr,
       nullptr, qb, M, D_MODEL, D_MODEL, D_MODEL, 0);
  gemm(c, Xh, D_MODEL, c.W(pi + 2), D_MODEL, c.F(pi + 3), nullptr, 0, nullptr,
       nullptr, kb, M, D_MODEL, D_MODEL, D_MODEL, 0);
  gemm(c, Xh, D_MODEL, c.W(pi + 4), D_MODEL, c.F(pi + 5), nullptr, 0, nullptr,
       nullptr, vb, M, D_MODEL, D_MODEL, D_MODEL, 0);
  k_attn<<<BATCH * NHEAD, 256, 0, c.stream>>>(qb, kb, vb, ab, valid, 0.125f);
  gemm(c, ab, D_MODEL, c.W(pi + 6), D_MODEL, c.F(pi + 7), res, D_MODEL, nullptr,
       outF, outH, M, D_MODEL, D_MODEL, D_MODEL, 0);
}

extern "C" void kernel_launch(void* const* d_in, const int* in_sizes, int n_in,
                              void* d_out, int out_size, void* d_ws, size_t ws_size,
                              hipStream_t stream) {
  Ctx c{stream, d_in, in_sizes, n_in, (char*)d_ws, 0, ws_size};
  for (int i = 0; i < 512; ++i) c.Wb[i] = nullptr;

  // Convert every parameter leaf to bf16 (deterministic, every call).
  int np = n_in < 512 ? n_in : 512;
  for (int i = P0; i < np; ++i) {
    size_t n = (size_t)in_sizes[i];
    __bf16* p = walloc<__bf16>(c, n);
    cvt(c, (const float*)d_in[i], p, n);
    c.Wb[i] = p;
  }

  const int    M1    = BATCH * NTOK;                 // 256 face tokens
  const size_t ME    = (size_t)BATCH * NTOK * NTOK;  // 16384 edge tokens
  const float* maskE = (const float*)d_in[IN_EMASK];
  const unsigned char* valid = (const unsigned char*)d_in[IN_FMASK];

  // ---- embeddings -------------------------------------------------------
  float* temb_in = walloc<float>(c, BATCH * D_MODEL);
  k_sincos<<<gblk(BATCH * (D_MODEL / 2)), 256, 0, stream>>>((const int*)d_in[IN_T], temb_in);
  __bf16* temb_in_h = walloc<__bf16>(c, BATCH * D_MODEL);
  cvt(c, temb_in, temb_in_h, (size_t)BATCH * D_MODEL);
  float* temb = walloc<float>(c, BATCH * D_MODEL);
  mlp4(c, temb_in_h, BATCH, D_MODEL, D_MODEL, D_MODEL, P_TIME,
       nullptr, nullptr, temb, nullptr, D_MODEL);

  __bf16* surfPos_h = walloc<__bf16>(c, (size_t)M1 * 6);
  cvt(c, (const float*)d_in[IN_SURFPOS], surfPos_h, (size_t)M1 * 6);
  __bf16* surfZ_h = walloc<__bf16>(c, (size_t)M1 * 48);
  cvt(c, (const float*)d_in[IN_SURFZ], surfZ_h, (size_t)M1 * 48);
  float* surf = walloc<float>(c, (size_t)M1 * D_MODEL);
  mlp4(c, surfPos_h, M1, 6, D_MODEL, D_MODEL, P_SURFP, nullptr, nullptr,
       surf, nullptr, D_MODEL);
  mlp4(c, surfZ_h, M1, 48, D_MODEL, D_MODEL, P_SURFZ, surf, nullptr,
       surf, nullptr, D_MODEL);                       // surf += surfz branch

  __bf16* edgePos_h = walloc<__bf16>(c, ME * 6);
  cvt(c, (const float*)d_in[IN_EDGEPOS], edgePos_h, ME * 6);
  float* eemb = walloc<float>(c, ME * D_MODEL);
  mlp4(c, edgePos_h, (int)ME, 6, D_MODEL, D_MODEL, P_EDGEP, nullptr, maskE,
       eemb, nullptr, D_MODEL);                       // eemb * edge_mask fused

  float* dez = walloc<float>(c, ME * D_MODEL);        // dense_ez (h_edge init)
  k_edge_tokens<<<gblk(ME * D_MODEL), 256, 0, stream>>>(surf, eemb, temb, maskE,
                                                        dez, nullptr);
  float* pooled = walloc<float>(c, (size_t)M1 * D_MODEL);
  k_pool_mean<<<gblk((size_t)M1 * D_MODEL), 256, 0, stream>>>(eemb, pooled);
  float* x     = walloc<float>(c, (size_t)M1 * D_MODEL);
  float* hface = walloc<float>(c, (size_t)M1 * D_MODEL);
  k_face_init<<<gblk((size_t)M1 * D_MODEL), 256, 0, stream>>>(surf, pooled, temb, x, hface);

  // ---- 12 transformer encoder layers ------------------------------------
  __bf16* xh  = walloc<__bf16>(c, (size_t)M1 * D_MODEL);
  __bf16* qb  = walloc<__bf16>(c, (size_t)M1 * D_MODEL);
  __bf16* kb  = walloc<__bf16>(c, (size_t)M1 * D_MODEL);
  __bf16* vb  = walloc<__bf16>(c, (size_t)M1 * D_MODEL);
  __bf16* ab  = walloc<__bf16>(c, (size_t)M1 * D_MODEL);
  float*  x2  = walloc<float>(c, (size_t)M1 * D_MODEL);
  __bf16* hh  = walloc<__bf16>(c, (size_t)M1 * D_MODEL);
  __bf16* h1h = walloc<__bf16>(c, (size_t)M1 * 1024);
  for (int L = 0; L < 12; ++L) {
    int pb = P_NET + L * 16;
    k_layernorm<<<M1, 256, 0, stream>>>(x, c.F(pb + 12), c.F(pb + 13),
                                        nullptr, xh, D_MODEL, 0);
    mha(c, xh, pb, x, x2, nullptr, qb, kb, vb, ab, valid);
    k_layernorm<<<M1, 256, 0, stream>>>(x2, c.F(pb + 14), c.F(pb + 15),
                                        nullptr, hh, D_MODEL, 0);
    gemm(c, hh, D_MODEL, c.W(pb + 8), 1024, c.F(pb + 9), nullptr, 0, nullptr,
         nullptr, h1h, M1, 1024, D_MODEL, 1024, 1 /*relu*/);
    gemm(c, h1h, 1024, c.W(pb + 10), D_MODEL, c.F(pb + 11), x2, D_MODEL, nullptr,
         x, nullptr, M1, D_MODEL, 1024, D_MODEL, 0);
  }
  float* gsurf = walloc<float>(c, (size_t)M1 * D_MODEL);
  k_layernorm<<<M1, 256, 0, stream>>>(x, c.F(P_NETLN), c.F(P_NETLN + 1),
                                      gsurf, nullptr, D_MODEL, 0);

  // ---- 6 LTIA (GINE + global attention) blocks --------------------------
  float*  fh   = walloc<float>(c, (size_t)M1 * D_MODEL);  // 6 x 128 face hids
  float*  eh   = walloc<float>(c, ME * D_MODEL);          // 6 x 128 edge hids
  __bf16* hfh  = walloc<__bf16>(c, (size_t)M1 * D_MODEL);
  __bf16* hfnh = walloc<__bf16>(c, (size_t)M1 * D_MODEL);
  __bf16* hlh  = walloc<__bf16>(c, (size_t)M1 * D_MODEL);
  __bf16* g1h  = walloc<__bf16>(c, (size_t)M1 * D_MODEL);
  float*  hl2  = walloc<float>(c, (size_t)M1 * D_MODEL);
  float*  tsum = walloc<float>(c, (size_t)M1 * D_MODEL);
  __bf16* eph  = walloc<__bf16>(c, ME * D_MODEL);
  float*  heA  = walloc<float>(c, ME * D_MODEL);
  float*  heB  = walloc<float>(c, ME * D_MODEL);
  __bf16* heH  = walloc<__bf16>(c, ME * D_MODEL);
  float* hedge = dez;                                     // h_edge state
  for (int G = 0; G < 6; ++G) {
    int pb = P_GNN + G * 19;
    float* hnew = (G & 1) ? heB : heA;
    // GINE aggregation -> bf16 input of the update MLP
    k_gine<<<gblk((size_t)M1 * D_MODEL), 256, 0, stream>>>(hface, hedge, maskE,
                                                           c.F(pb + 0), hlh);
    gemm(c, hlh, D_MODEL, c.W(pb + 1), D_MODEL, c.F(pb + 2), nullptr, 0, nullptr,
         nullptr, g1h, M1, D_MODEL, D_MODEL, D_MODEL, 1 /*relu*/);
    gemm(c, g1h, D_MODEL, c.W(pb + 3), D_MODEL, c.F(pb + 4), nullptr, 0, nullptr,
         hl2, nullptr, M1, D_MODEL, D_MODEL, D_MODEL, 0);
    // global attention branch on raw h_face; new h_face = h_face + hl + hg
    cvt(c, hface, hfh, (size_t)M1 * D_MODEL);
    k_add2<<<gblk((size_t)M1 * D_MODEL), 256, 0, stream>>>(hface, hl2, tsum,
                                                           (size_t)M1 * D_MODEL);
    mha(c, hfh, pb + 5, tsum, hface, hfnh, qb, kb, vb, ab, valid);
    // face hidden projection into fh columns [G*128, G*128+128)
    gemm(c, hfnh, D_MODEL, c.W(pb + 15), 128, c.F(pb + 16), nullptr, 0, nullptr,
         fh + G * 128, nullptr, M1, 128, D_MODEL, D_MODEL, 0);
    // edge update: silu(ew(he + hf_i + hf_j)) * adj
    k_edge_pair<<<gblk(ME * D_MODEL), 256, 0, stream>>>(hface, hedge, eph);
    gemm(c, eph, D_MODEL, c.W(pb + 13), D_MODEL, c.F(pb + 14), nullptr, 0, maskE,
         hnew, heH, (int)ME, D_MODEL, D_MODEL, D_MODEL, 2 /*silu*/);
    // edge hidden projection into eh columns
    gemm(c, heH, D_MODEL, c.W(pb + 17), 128, c.F(pb + 18), nullptr, 0, nullptr,
         eh + G * 128, nullptr, (int)ME, 128, D_MODEL, D_MODEL, 0);
    hedge = hnew;
  }

  // ---- output head ------------------------------------------------------
  __bf16* cat_h = walloc<__bf16>(c, ME * 2304);
  k_concat<<<gblk(ME * 2304), 256, 0, stream>>>(dez, eh, fh, maskE, cat_h);
  float* ep = walloc<float>(c, ME * D_MODEL);
  mlp4(c, cat_h, (int)ME, 2304, D_MODEL, D_MODEL, P_FCOUT2, nullptr, nullptr,
       ep, nullptr, D_MODEL);
  __bf16* ep2h = walloc<__bf16>(c, ME * D_MODEL);
  k_add_ge<<<gblk(ME * D_MODEL), 256, 0, stream>>>(ep, gsurf, maskE, ep2h);
  __bf16* ep3h = walloc<__bf16>(c, ME * D_MODEL);
  mlp4(c, ep2h, (int)ME, D_MODEL, D_MODEL, D_MODEL, P_FCPOOL, nullptr, nullptr,
       nullptr, ep3h, D_MODEL);
  float* pred = walloc<float>(c, ME * 6);
  mlp4(c, ep3h, (int)ME, D_MODEL, D_MODEL, 6, P_FCOUT, nullptr, maskE,
       pred, nullptr, 6);
  k_sym<<<gblk(ME * 6), 256, 0, stream>>>(pred, (float*)d_out);
  (void)out_size; (void)ws_size;
}
